// MultiHeadAttention_38938173505645
// MI455X (gfx1250) — matrix operand order repaired and checked by reading, not compiled, not run on hardware
//
#include <hip/hip_runtime.h>
#include <hip/hip_bf16.h>
#include <stdint.h>

// Problem constants (from reference)
#define B_  2
#define S_  2048
#define E_  1024
#define H_  16
#define DH_ 64

// Padded LDS row strides (elements) -> 16 disjoint 4-bank groups per frag read
#define KSTR 72   // K tile rows (64 + 8)
#define VSTR 40   // V^T / W tile rows (32 + 8)
#define PSTR 40   // P tile rows (32 + 8)

typedef __bf16 bf16;
typedef __attribute__((ext_vector_type(16))) __bf16   v16bf;
typedef __attribute__((ext_vector_type(8)))  float    v8f;
typedef __attribute__((ext_vector_type(4)))  uint32_t u32x4;
typedef __attribute__((ext_vector_type(4)))  int      v4i;
typedef __attribute__((ext_vector_type(4)))  float    f32x4;

// Address-space-qualified int4 pointers for the gfx1250 async-to-LDS builtin
typedef __attribute__((address_space(1))) v4i* gptr_v4i;
typedef __attribute__((address_space(3))) v4i* lptr_v4i;

// Fragment container: v16bf = 32 bytes = two 128-bit loads
union ABFrag { v16bf v; u32x4 q[2]; };

#if __has_builtin(__builtin_amdgcn_global_load_async_to_lds_b128) && \
    __has_builtin(__builtin_amdgcn_s_wait_asynccnt)
#define HAS_ASYNC_LDS 1
#else
#define HAS_ASYNC_LDS 0
#endif

// 16-byte global -> LDS copy.  Async (ASYNCcnt-tracked, bypasses VGPRs) when
// the gfx1250 async builtins exist; otherwise plain load+ds_store fallback.
// Generic->AS1 is a numeric identity; generic->AS3 truncates to the low 32
// bits (a flat LDS address maps to its low 32 bits).
__device__ __forceinline__ void cp_g2l_16(void* lds, const void* g) {
#if HAS_ASYNC_LDS
  __builtin_amdgcn_global_load_async_to_lds_b128(
      (gptr_v4i)(uintptr_t)g,
      (lptr_v4i)(uint32_t)(uintptr_t)lds,
      0, 0);
#else
  *(u32x4*)lds = *(const u32x4*)g;
#endif
}
__device__ __forceinline__ void cp_g2l_wait() {
#if HAS_ASYNC_LDS
  __builtin_amdgcn_s_wait_asynccnt(0);
#endif
}

__device__ __forceinline__ bf16 f2bf(float f) {
  union { float f; uint32_t u; } x; x.f = f;
  uint32_t r = x.u + 0x7FFFu + ((x.u >> 16) & 1u);  // round-to-nearest-even
  union { uint16_t u; bf16 b; } y; y.u = (uint16_t)(r >> 16);
  return y.b;
}

__device__ __forceinline__ v8f wmma_bf16(const ABFrag& a, const ABFrag& b, v8f c) {
  return __builtin_amdgcn_wmma_f32_16x16x32_bf16(false, a.v, false, b.v,
                                                 (short)0, c, false, false);
}

// ---------------------------------------------------------------------------
// Conversion kernels
// ---------------------------------------------------------------------------
__global__ void cvt_f32_bf16(const float* __restrict__ in, bf16* __restrict__ out, int n) {
  int i = (blockIdx.x * blockDim.x + threadIdx.x) * 4;
  int stride = gridDim.x * blockDim.x * 4;
  for (; i < n; i += stride) {
    f32x4 v = *(const f32x4*)(in + i);
    out[i + 0] = f2bf(v.x); out[i + 1] = f2bf(v.y);
    out[i + 2] = f2bf(v.z); out[i + 3] = f2bf(v.w);
  }
}

// W [H,E,DH] (f32) -> Wt [H,DH,E] (bf16), per-head transpose
__global__ void cvt_w_headT(const float* __restrict__ w, bf16* __restrict__ wt) {
  int i = blockIdx.x * blockDim.x + threadIdx.x;  // over H*E*DH
  if (i >= H_ * E_ * DH_) return;
  int d = i % DH_;
  int e = (i / DH_) % E_;
  int h = i / (DH_ * E_);
  wt[((size_t)h * DH_ + d) * E_ + e] = f2bf(w[i]);
}

// Wo [E,E] (f32) -> Wot [E,E] (bf16) transposed
__global__ void cvt_woT(const float* __restrict__ w, bf16* __restrict__ wt) {
  int i = blockIdx.x * blockDim.x + threadIdx.x;  // i = e*E + n
  if (i >= E_ * E_) return;
  int n = i % E_;
  int e = i / E_;
  wt[(size_t)n * E_ + e] = f2bf(w[i]);
}

// ---------------------------------------------------------------------------
// Per-head projection: out[b,h] = (X[b] @ W[h] + bias[h]) * scale
// One wave computes a 32x64 output tile (8 WMMA per k-step).  The shared
// 64x32 W k-slice is async-copied into double-buffered LDS by the whole
// block (16 B/thread); A-fragments are register double-buffered from global.
// ---------------------------------------------------------------------------
__global__ __launch_bounds__(256) void qkv_proj(
    const bf16* __restrict__ X, const bf16* __restrict__ Wt,
    const float* __restrict__ bias, bf16* __restrict__ out,
    float scale, int vtrans) {
  __shared__ bf16 Wl[2][64 * VSTR];

  int tid = threadIdx.x;
  int lane = tid & 31, wave = tid >> 5;
  int bb = blockIdx.z, h = blockIdx.y;
  int rowBase = blockIdx.x * 256 + wave * 32;
  int half = (lane < 16) ? 0 : 1;
  int col  = lane & 15;
  int aOff = half ? 16 : 0;   // A-frag K half select   (contiguous k = 16h + s)
  int bOff = half ? 16 : 0;   // B-frag K half select   (contiguous k = 16h + s, same as A)

  const bf16* Xp = X + ((size_t)(bb * S_ + rowBase)) * E_;
  const bf16* Wp = Wt + (size_t)h * DH_ * E_;
  const bf16* ar0 = Xp + col * E_ + aOff;
  const bf16* ar1 = Xp + (16 + col) * E_ + aOff;

  // W staging map: thread -> (row d = tid/4, 16B chunk (tid%4)*8) of 64x32 slice
  int wrow = tid >> 2, woff = (tid & 3) * 8;
  const bf16* Wg = Wp + (size_t)wrow * E_ + woff;
  int wlds = wrow * VSTR + woff;

  ABFrag af[2][2];
  auto loadA = [&](int buf, int k) {
    af[buf][0].q[0] = *(const u32x4*)(ar0 + k);
    af[buf][0].q[1] = *(const u32x4*)(ar0 + k + 8);
    af[buf][1].q[0] = *(const u32x4*)(ar1 + k);
    af[buf][1].q[1] = *(const u32x4*)(ar1 + k + 8);
  };

  // prologue: fill LDS slice for k=0, load A-frags for k=0
  cp_g2l_16(&Wl[0][wlds], Wg);
  loadA(0, 0);
  cp_g2l_wait();
  __syncthreads();

  v8f acc[2][4] = {};
  for (int k = 0; k < E_; k += 32) {
    int cur = (k >> 5) & 1;
    bool more = (k + 32) < E_;
    if (more) {                           // issue next-slice copies early
      cp_g2l_16(&Wl[cur ^ 1][wlds], Wg + k + 32);
      loadA(cur ^ 1, k + 32);
    }
    // B-frags from LDS (padded rows -> conflict-free b128 reads)
    ABFrag bfr[4];
#pragma unroll
    for (int nt = 0; nt < 4; ++nt) {
      const bf16* bp = &Wl[cur][(nt * 16 + col) * VSTR + bOff];
      bfr[nt].q[0] = *(const u32x4*)(bp);
      bfr[nt].q[1] = *(const u32x4*)(bp + 8);
    }
#pragma unroll
    for (int nt = 0; nt < 4; ++nt) {
      acc[0][nt] = wmma_bf16(af[cur][0], bfr[nt], acc[0][nt]);
      acc[1][nt] = wmma_bf16(af[cur][1], bfr[nt], acc[1][nt]);
    }
    cp_g2l_wait();
    __syncthreads();
  }

  const float* bp = bias + h * DH_;
#pragma unroll
  for (int mi = 0; mi < 2; ++mi) {
#pragma unroll
    for (int nt = 0; nt < 4; ++nt) {
      float bv = bp[nt * 16 + col];
#pragma unroll
      for (int j = 0; j < 8; ++j) {
        int row = rowBase + mi * 16 + j + half * 8;
        float v = (acc[mi][nt][j] + bv) * scale;
        if (!vtrans)
          out[(((size_t)(bb * H_ + h)) * S_ + row) * DH_ + nt * 16 + col] = f2bf(v);
        else
          out[(((size_t)(bb * H_ + h)) * DH_ + nt * 16 + col) * S_ + row] = f2bf(v);
      }
    }
  }
}

// ---------------------------------------------------------------------------
// Flash attention: block = 8 waves x 16-row Q strips, one (b,h).
// K tile (32x64) and V^T tile (64x32) are async-copied into double-buffered
// LDS by the whole block (one 16B chunk per thread); the copy for tile n+1 is
// issued at the top of iteration n and completes under the WMMAs + softmax;
// a single s_wait_asynccnt + barrier closes each iteration.
// ---------------------------------------------------------------------------
__global__ __launch_bounds__(256) void attn(
    const bf16* __restrict__ Q, const bf16* __restrict__ K,
    const bf16* __restrict__ Vt, const int* __restrict__ mask,
    bf16* __restrict__ CTX) {
  __shared__ bf16 Kl[2][32 * KSTR];
  __shared__ bf16 Vl[2][64 * VSTR];
  __shared__ bf16 Pl[8][16 * PSTR];

  int tid = threadIdx.x;
  int lane = tid & 31, wave = tid >> 5;
  int bb = blockIdx.z, h = blockIdx.y;
  int rowBase = blockIdx.x * 128 + wave * 16;
  int half = (lane < 16) ? 0 : 1;
  int col  = lane & 15;
  int aOff = half ? 16 : 0;
  int bOff = half ? 16 : 0;

  const bf16* Qp = Q + (((size_t)(bb * H_ + h)) * S_ + rowBase) * DH_;
  const bf16* Kp = K + ((size_t)(bb * H_ + h)) * S_ * DH_;
  const bf16* Vp = Vt + ((size_t)(bb * H_ + h)) * DH_ * S_;
  const int*  Mp = mask + (size_t)bb * S_ * S_;

  // staging maps: K: thread -> (key row tid/8, chunk (tid%8)*8) of 32x64
  //               V: thread -> (d row tid/4,   chunk (tid%4)*8) of 64x32
  int krow = tid >> 3, koff = (tid & 7) * 8;
  int vrow = tid >> 2, voff = (tid & 3) * 8;
  const bf16* Kg = Kp + (size_t)krow * DH_ + koff;
  const bf16* Vg = Vp + (size_t)vrow * S_ + voff;
  int klds = krow * KSTR + koff;
  int vlds = vrow * VSTR + voff;

  // Q A-fragments, kept resident (d-steps 0 and 1 of DH=64)
  ABFrag qf0, qf1;
  {
    const bf16* p = Qp + col * DH_ + aOff;
    qf0.q[0] = *(const u32x4*)(p);      qf0.q[1] = *(const u32x4*)(p + 8);
    qf1.q[0] = *(const u32x4*)(p + 32); qf1.q[1] = *(const u32x4*)(p + 40);
  }

  // prologue: fill tile 0
  cp_g2l_16(&Kl[0][klds], Kg);
  cp_g2l_16(&Vl[0][vlds], Vg);
  cp_g2l_wait();
  __syncthreads();

  v8f acc[4] = {};
  float mrow[8], lrow[8];
#pragma unroll
  for (int j = 0; j < 8; ++j) { mrow[j] = -3.0e38f; lrow[j] = 0.0f; }

  bf16* Pw = &Pl[wave][0];

  for (int t = 0; t < S_; t += 32) {
    int cur = (t >> 5) & 1;
    bool more = (t + 32) < S_;

    // ---- issue next tile's async copies + mask loads early ----
    if (more) {
      cp_g2l_16(&Kl[cur ^ 1][klds], Kg + (size_t)(t + 32) * DH_);
      cp_g2l_16(&Vl[cur ^ 1][vlds], Vg + t + 32);
    }
    int m0[8], m1[8];
#pragma unroll
    for (int j = 0; j < 8; ++j) {
      const int* mrp = Mp + (size_t)(rowBase + j + half * 8) * S_ + t + col;
      m0[j] = mrp[0]; m1[j] = mrp[16];
    }

    // ---- K B-frags from LDS, then score WMMAs ----
    ABFrag kf[4];
    {
      const bf16* p  = &Kl[cur][col * KSTR + bOff];
      const bf16* p2 = &Kl[cur][(16 + col) * KSTR + bOff];
      kf[0].q[0] = *(const u32x4*)(p);       kf[0].q[1] = *(const u32x4*)(p + 8);
      kf[1].q[0] = *(const u32x4*)(p + 32);  kf[1].q[1] = *(const u32x4*)(p + 40);
      kf[2].q[0] = *(const u32x4*)(p2);      kf[2].q[1] = *(const u32x4*)(p2 + 8);
      kf[3].q[0] = *(const u32x4*)(p2 + 32); kf[3].q[1] = *(const u32x4*)(p2 + 40);
    }
    v8f c0 = {}, c1 = {};
    c0 = wmma_bf16(qf0, kf[0], c0);
    c0 = wmma_bf16(qf1, kf[1], c0);
    c1 = wmma_bf16(qf0, kf[2], c1);
    c1 = wmma_bf16(qf1, kf[3], c1);

    // ---- V B-frags from LDS into independent registers (clause + partial waits)
    ABFrag vfr[4];
#pragma unroll
    for (int nt = 0; nt < 4; ++nt) {
      const bf16* p = &Vl[cur][(nt * 16 + col) * VSTR + bOff];
      vfr[nt].q[0] = *(const u32x4*)(p);
      vfr[nt].q[1] = *(const u32x4*)(p + 8);
    }

    // ---- mask + online softmax (rows live across 16-lane halves) ----
#pragma unroll
    for (int j = 0; j < 8; ++j) {
      float s0 = m0[j] ? c0[j] : -3.0e38f;
      float s1 = m1[j] ? c1[j] : -3.0e38f;
      float mx = fmaxf(s0, s1);
#pragma unroll
      for (int off = 1; off < 16; off <<= 1) mx = fmaxf(mx, __shfl_xor(mx, off, 32));
      float newm = fmaxf(mrow[j], mx);
      float corr = __expf(mrow[j] - newm);
      float p0 = m0[j] ? __expf(s0 - newm) : 0.0f;
      float p1 = m1[j] ? __expf(s1 - newm) : 0.0f;
      float rs = p0 + p1;
#pragma unroll
      for (int off = 1; off < 16; off <<= 1) rs += __shfl_xor(rs, off, 32);
      lrow[j] = lrow[j] * corr + rs;
      mrow[j] = newm;
      acc[0][j] *= corr; acc[1][j] *= corr; acc[2][j] *= corr; acc[3][j] *= corr;
      int m = j + half * 8;                 // C-layout -> LDS row-major P tile
      Pw[m * PSTR + col]      = f2bf(p0);
      Pw[m * PSTR + 16 + col] = f2bf(p1);
    }

    // ---- re-load P as A-fragment (same-wave DS ordering is in-order) ----
    ABFrag pf;
    {
      const bf16* p = Pw + col * PSTR + aOff;
      pf.q[0] = *(const u32x4*)(p);
      pf.q[1] = *(const u32x4*)(p + 8);
    }

    // ---- ctx += P @ V[t..t+31] ----
#pragma unroll
    for (int nt = 0; nt < 4; ++nt)
      acc[nt] = wmma_bf16(pf, vfr[nt], acc[nt]);

    cp_g2l_wait();
    __syncthreads();
  }

  // ---- normalize and emit concatenated-head context, bf16 ----
#pragma unroll
  for (int j = 0; j < 8; ++j) {
    float inv = 1.0f / lrow[j];
    int row = rowBase + j + half * 8;
    bf16* op = CTX + ((size_t)(bb * S_ + row)) * E_ + h * DH_;
#pragma unroll
    for (int nt = 0; nt < 4; ++nt) op[nt * 16 + col] = f2bf(acc[nt][j] * inv);
  }
}

// ---------------------------------------------------------------------------
// Output projection: out = CTX @ Wo + bo   (f32 output)
// Same LDS-staged, double-buffered structure as qkv_proj.
// ---------------------------------------------------------------------------
__global__ __launch_bounds__(256) void out_proj(
    const bf16* __restrict__ CTX, const bf16* __restrict__ Wot,
    const float* __restrict__ bo, float* __restrict__ out) {
  __shared__ bf16 Wl[2][64 * VSTR];

  int tid = threadIdx.x;
  int lane = tid & 31, wave = tid >> 5;
  int bb = blockIdx.z;
  int n0 = blockIdx.y * 64;
  int rowBase = blockIdx.x * 256 + wave * 32;
  int half = (lane < 16) ? 0 : 1;
  int col  = lane & 15;
  int aOff = half ? 16 : 0;
  int bOff = half ? 16 : 0;

  const bf16* Xp = CTX + ((size_t)(bb * S_ + rowBase)) * E_;
  const bf16* ar0 = Xp + col * E_ + aOff;
  const bf16* ar1 = Xp + (16 + col) * E_ + aOff;

  int wrow = tid >> 2, woff = (tid & 3) * 8;
  const bf16* Wg = Wot + (size_t)(n0 + wrow) * E_ + woff;
  int wlds = wrow * VSTR + woff;

  ABFrag af[2][2];
  auto loadA = [&](int buf, int k) {
    af[buf][0].q[0] = *(const u32x4*)(ar0 + k);
    af[buf][0].q[1] = *(const u32x4*)(ar0 + k + 8);
    af[buf][1].q[0] = *(const u32x4*)(ar1 + k);
    af[buf][1].q[1] = *(const u32x4*)(ar1 + k + 8);
  };

  cp_g2l_16(&Wl[0][wlds], Wg);
  loadA(0, 0);
  cp_g2l_wait();
  __syncthreads();

  v8f acc[2][4] = {};
  for (int k = 0; k < E_; k += 32) {
    int cur = (k >> 5) & 1;
    bool more = (k + 32) < E_;
    if (more) {
      cp_g2l_16(&Wl[cur ^ 1][wlds], Wg + k + 32);
      loadA(cur ^ 1, k + 32);
    }
    ABFrag bfr[4];
#pragma unroll
    for (int nt = 0; nt < 4; ++nt) {
      const bf16* bp = &Wl[cur][(nt * 16 + col) * VSTR + bOff];
      bfr[nt].q[0] = *(const u32x4*)(bp);
      bfr[nt].q[1] = *(const u32x4*)(bp + 8);
    }
#pragma unroll
    for (int nt = 0; nt < 4; ++nt) {
      acc[0][nt] = wmma_bf16(af[cur][0], bfr[nt], acc[0][nt]);
      acc[1][nt] = wmma_bf16(af[cur][1], bfr[nt], acc[1][nt]);
    }
    cp_g2l_wait();
    __syncthreads();
  }

#pragma unroll
  for (int mi = 0; mi < 2; ++mi) {
#pragma unroll
    for (int nt = 0; nt < 4; ++nt) {
      float bv = bo[n0 + nt * 16 + col];
#pragma unroll
      for (int j = 0; j < 8; ++j) {
        int row = rowBase + mi * 16 + j + half * 8;
        out[((size_t)(bb * S_ + row)) * E_ + n0 + nt * 16 + col] = acc[mi][nt][j] + bv;
      }
    }
  }
}

// ---------------------------------------------------------------------------
extern "C" void kernel_launch(void* const* d_in, const int* in_sizes, int n_in,
                              void* d_out, int out_size, void* d_ws, size_t ws_size,
                              hipStream_t stream) {
  (void)in_sizes; (void)n_in; (void)out_size; (void)ws_size;
  const float* hidden = (const float*)d_in[0];
  const int*   mask   = (const int*)d_in[1];
  const float* Wq = (const float*)d_in[2];
  const float* bq = (const float*)d_in[3];
  const float* Wk = (const float*)d_in[4];
  const float* bk = (const float*)d_in[5];
  const float* Wv = (const float*)d_in[6];
  const float* bv = (const float*)d_in[7];
  const float* Wo = (const float*)d_in[8];
  const float* bo = (const float*)d_in[9];
  float* outp = (float*)d_out;

  // Workspace carve-out (~48 MB total), 256B aligned
  char* ws = (char*)d_ws;
  size_t off = 0;
  auto alloc = [&](size_t bytes) -> void* {
    void* p = ws + off;
    off += (bytes + 255) & ~(size_t)255;
    return p;
  };
  bf16* Xbf  = (bf16*)alloc((size_t)B_ * S_ * E_ * 2);
  bf16* Wqt  = (bf16*)alloc((size_t)H_ * DH_ * E_ * 2);
  bf16* Wkt  = (bf16*)alloc((size_t)H_ * DH_ * E_ * 2);
  bf16* Wvt  = (bf16*)alloc((size_t)H_ * DH_ * E_ * 2);
  bf16* Wot  = (bf16*)alloc((size_t)E_ * E_ * 2);
  bf16* Qb   = (bf16*)alloc((size_t)B_ * H_ * S_ * DH_ * 2);
  bf16* Kb   = (bf16*)alloc((size_t)B_ * H_ * S_ * DH_ * 2);
  bf16* Vtb  = (bf16*)alloc((size_t)B_ * H_ * S_ * DH_ * 2);
  bf16* CTXb = (bf16*)alloc((size_t)B_ * S_ * E_ * 2);

  // 1) hidden -> bf16
  cvt_f32_bf16<<<1024, 256, 0, stream>>>(hidden, Xbf, B_ * S_ * E_);
  // 2) weights -> bf16, transposed for B-fragment-contiguous loads
  int nw = H_ * E_ * DH_;
  cvt_w_headT<<<(nw + 255) / 256, 256, 0, stream>>>(Wq, Wqt);
  cvt_w_headT<<<(nw + 255) / 256, 256, 0, stream>>>(Wk, Wkt);
  cvt_w_headT<<<(nw + 255) / 256, 256, 0, stream>>>(Wv, Wvt);
  cvt_woT<<<(E_ * E_ + 255) / 256, 256, 0, stream>>>(Wo, Wot);

  // 3) Q/K/V projections (scale 1/sqrt(64) folded into Q; V stored transposed)
  dim3 gp(S_ / 256, H_, B_);
  qkv_proj<<<gp, 256, 0, stream>>>(Xbf, Wqt, bq, Qb, 0.125f, 0);
  qkv_proj<<<gp, 256, 0, stream>>>(Xbf, Wkt, bk, Kb, 1.0f, 0);
  qkv_proj<<<gp, 256, 0, stream>>>(Xbf, Wvt, bv, Vtb, 1.0f, 1);

  // 4) flash attention
  dim3 ga(S_ / 128, H_, B_);
  attn<<<ga, 256, 0, stream>>>(Qb, Kb, Vtb, mask, CTXb);

  // 5) output projection -> f32
  dim3 go(S_ / 256, E_ / 64, B_);
  out_proj<<<go, 256, 0, stream>>>(CTXb, Wot, bo, outp);
}
